// ChebConvNet_8074538516512
// MI455X (gfx1250) — compile-verified, matrix operand order checked by reading, not run on hardware
//
#include <hip/hip_runtime.h>
#include <hip/hip_bf16.h>

typedef __attribute__((ext_vector_type(16))) _Float16 v16h;
typedef __attribute__((ext_vector_type(8)))  _Float16 v8h;
typedef __attribute__((ext_vector_type(4)))  _Float16 v4h;
typedef __attribute__((ext_vector_type(8)))  float    v8f;

#define N_NODES    200000
#define D_FEAT     128
#define N_CLASSES  40
#define CPAD       48          // classes padded to 3 WMMA N-tiles
#define PITCH      136         // halfs per LDS row (128 + 8 pad -> bank-conflict free b128)
#define BLOCK_ROWS 128         // 8 waves x 16 rows
#define THREADS    256
#define N_BLOCKS   ((N_NODES + BLOCK_ROWS - 1) / BLOCK_ROWS)   // 1563
#define GRID_WG    640

// Load a 16x32 f16 A/B fragment for one lane from an LDS row (row-major, PITCH stride).
// Per the CDNA5 VGPR layout: halfwords 0..7 = K[half*8 .. +8), 8..15 = K[16+half*8 .. +8).
__device__ __forceinline__ v16h load_frag(const _Float16* __restrict__ row, int kbase) {
  v8h lo = *(const v8h*)(row + kbase);
  v8h hi = *(const v8h*)(row + kbase + 16);
  return __builtin_shufflevector(lo, hi, 0,1,2,3,4,5,6,7,8,9,10,11,12,13,14,15);
}

// Fast SiLU: v * sigmoid(v) with single-instruction v_rcp_f32 (no IEEE div expansion).
__device__ __forceinline__ float fast_silu(float v) {
  return v * __builtin_amdgcn_rcpf(1.0f + __expf(-v));
}

__global__ __launch_bounds__(THREADS) void cheb_mlp_wmma_kernel(
    const float* __restrict__ x,
    const float* __restrict__ W0, const float* __restrict__ b0,
    const float* __restrict__ W1, const float* __restrict__ b1,
    const float* __restrict__ W2, const float* __restrict__ b2,
    const float* __restrict__ W3, const float* __restrict__ b3,
    float* __restrict__ out)
{
  extern __shared__ char smem[];
  _Float16* sW  = (_Float16*)smem;                    // [3][128][PITCH]  W^T in f16
  _Float16* sW3 = sW  + 3 * D_FEAT * PITCH;           // [48][PITCH]      W3^T padded
  _Float16* sH  = sW3 + CPAD * PITCH;                 // [128][PITCH]     activations
  float*    sL  = (float*)(sH + BLOCK_ROWS * PITCH);  // [128][48]        logits
  float*    sB  = sL + BLOCK_ROWS * CPAD;             // [3*128 + 48]     biases

  const int tid  = threadIdx.x;
  const int wave = tid >> 5;
  const int lane = tid & 31;
  const int lo16 = lane & 15;
  const int h8   = (lane >> 4) << 3;   // 0 or 8 (which K/M half this lane covers)

  // ---- one-time per workgroup: transpose weights into LDS as f16, stage biases ----
  const float* Ws[3] = {W0, W1, W2};
  for (int l = 0; l < 3; ++l) {
    const float* W = Ws[l];
    _Float16* dst = sW + l * D_FEAT * PITCH;
    for (int e = tid; e < D_FEAT * D_FEAT; e += THREADS) {
      int k = e >> 7, n = e & 127;                 // W is [K][N] row-major
      dst[n * PITCH + k] = (_Float16)W[e];         // store W^T
    }
  }
  for (int e = tid; e < D_FEAT * N_CLASSES; e += THREADS) {
    int k = e / N_CLASSES, n = e - k * N_CLASSES;
    sW3[n * PITCH + k] = (_Float16)W3[e];
  }
  for (int e = tid; e < (CPAD - N_CLASSES) * D_FEAT; e += THREADS) {
    int n = N_CLASSES + (e >> 7), k = e & 127;
    sW3[n * PITCH + k] = (_Float16)0.0f;           // zero the padded classes
  }
  if (tid < D_FEAT) {                               // biases -> LDS (removes flat loads
    sB[0 * D_FEAT + tid] = b0[tid];                 //  from the WMMA inner loops)
    sB[1 * D_FEAT + tid] = b1[tid];
    sB[2 * D_FEAT + tid] = b2[tid];
  }
  if (tid < CPAD) sB[3 * D_FEAT + tid] = (tid < N_CLASSES) ? b3[tid] : 0.0f;
  __syncthreads();

  for (int blk = blockIdx.x; blk < N_BLOCKS; blk += (int)gridDim.x) {
    const int rowbase = blk * BLOCK_ROWS;

    // ---- stage this wave's 16 rows of x into sH as f16 (coalesced b128 loads) ----
    #pragma unroll 4
    for (int i = 0; i < 16; ++i) {
      int grow = rowbase + wave * 16 + i;
      grow = grow < N_NODES ? grow : (N_NODES - 1);   // clamp tail (results masked at store)
      const float4 v = *(const float4*)(x + (size_t)grow * D_FEAT + lane * 4);
      v4h hv; hv[0] = (_Float16)v.x; hv[1] = (_Float16)v.y;
              hv[2] = (_Float16)v.z; hv[3] = (_Float16)v.w;
      *(v4h*)(sH + (wave * 16 + i) * PITCH + lane * 4) = hv;
    }
    // each wave reads/writes only its own 16 rows of sH -> no barrier needed here

    const _Float16* arow = sH + (wave * 16 + lo16) * PITCH;
    const int rb = wave * 16 + h8;                    // accumulator row base for this lane

    // ---- 3 hidden layers: h = silu(h @ W + b), all WMMA f16->f32 ----
    #pragma unroll
    for (int l = 0; l < 3; ++l) {
      v16h af[4];
      #pragma unroll
      for (int kt = 0; kt < 4; ++kt) af[kt] = load_frag(arow, kt * 32 + h8);
      const _Float16* sWl = sW + l * D_FEAT * PITCH;
      const float*    sBl = sB + l * D_FEAT;
      #pragma unroll
      for (int nt = 0; nt < 8; ++nt) {
        const int n = nt * 16 + lo16;
        const float bv = sBl[n];                     // ds_load, overlaps with WMMA
        v8f acc = {bv, bv, bv, bv, bv, bv, bv, bv};
        const _Float16* brow = sWl + n * PITCH;
        #pragma unroll
        for (int kt = 0; kt < 4; ++kt) {
          v16h bf = load_frag(brow, kt * 32 + h8);
          acc = __builtin_amdgcn_wmma_f32_16x16x32_f16(
              false, af[kt], false, bf, (short)0, acc, false, false);
        }
        #pragma unroll
        for (int r = 0; r < 8; ++r) {
          sH[(rb + r) * PITCH + n] = (_Float16)fast_silu(acc[r]);  // write own rows only
        }
      }
    }

    // ---- head: logits = h @ W3 + b3 (3 N-tiles, padded) ----
    {
      v16h af[4];
      #pragma unroll
      for (int kt = 0; kt < 4; ++kt) af[kt] = load_frag(arow, kt * 32 + h8);
      #pragma unroll
      for (int nt = 0; nt < 3; ++nt) {
        const int n = nt * 16 + lo16;
        const float bv = sB[3 * D_FEAT + n];
        v8f acc = {bv, bv, bv, bv, bv, bv, bv, bv};
        const _Float16* brow = sW3 + n * PITCH;
        #pragma unroll
        for (int kt = 0; kt < 4; ++kt) {
          v16h bf = load_frag(brow, kt * 32 + h8);
          acc = __builtin_amdgcn_wmma_f32_16x16x32_f16(
              false, af[kt], false, bf, (short)0, acc, false, false);
        }
        #pragma unroll
        for (int r = 0; r < 8; ++r) sL[(rb + r) * CPAD + n] = acc[r];
      }
    }
    __syncthreads();   // logits complete across all waves

    // ---- per-row log_softmax over the 40 real classes ----
    if (tid < BLOCK_ROWS) {
      const int grow = rowbase + tid;
      if (grow < N_NODES) {
        const float* lr = sL + tid * CPAD;
        float m = -3.402823466e38f;
        #pragma unroll 8
        for (int j = 0; j < N_CLASSES; ++j) m = fmaxf(m, lr[j]);
        float s = 0.0f;
        #pragma unroll 8
        for (int j = 0; j < N_CLASSES; ++j) s += __expf(lr[j] - m);
        const float lse = m + __logf(s);
        float* op = out + (size_t)grow * N_CLASSES;
        #pragma unroll 8
        for (int j = 0; j < N_CLASSES; ++j) op[j] = lr[j] - lse;
      }
    }
    __syncthreads();   // protect sL (and sH reuse) before next block iteration
  }
}

extern "C" void kernel_launch(void* const* d_in, const int* in_sizes, int n_in,
                              void* d_out, int out_size, void* d_ws, size_t ws_size,
                              hipStream_t stream) {
  (void)in_sizes; (void)n_in; (void)out_size; (void)d_ws; (void)ws_size;
  const float* x  = (const float*)d_in[0];
  // d_in[1] = edge_index (int64) -- unused by the reference when K == 1
  const float* W0 = (const float*)d_in[2];
  const float* b0 = (const float*)d_in[3];
  const float* W1 = (const float*)d_in[4];
  const float* b1 = (const float*)d_in[5];
  const float* W2 = (const float*)d_in[6];
  const float* b2 = (const float*)d_in[7];
  const float* W3 = (const float*)d_in[8];
  const float* b3 = (const float*)d_in[9];
  float* out = (float*)d_out;

  const size_t smem_bytes =
      (size_t)(3 * D_FEAT * PITCH + CPAD * PITCH + BLOCK_ROWS * PITCH) * sizeof(_Float16)
      + (size_t)BLOCK_ROWS * CPAD * sizeof(float)
      + (size_t)(3 * D_FEAT + CPAD) * sizeof(float);   // ~178.6 KB (< 320 KB/WGP)

  int grid = GRID_WG < N_BLOCKS ? GRID_WG : N_BLOCKS;
  cheb_mlp_wmma_kernel<<<grid, THREADS, smem_bytes, stream>>>(
      x, W0, b0, W1, b1, W2, b2, W3, b3, out);
}
